// TransformerBlock_30777735643374
// MI455X (gfx1250) — matrix operand order repaired and checked by reading, not compiled, not run on hardware
//
#include <hip/hip_runtime.h>
#include <hip/hip_bf16.h>

typedef __attribute__((ext_vector_type(16))) _Float16 v16h;
typedef __attribute__((ext_vector_type(8)))  _Float16 v8h;
typedef __attribute__((ext_vector_type(8)))  float    v8f;
typedef int v4i __attribute__((vector_size(16)));                 // matches builtin's V4i
typedef __attribute__((address_space(1))) v4i* gptr_v4i;          // global (AS1)
typedef __attribute__((address_space(3))) v4i* lptr_v4i;          // LDS (AS3, 32-bit)

// Problem constants
static constexpr int Bz   = 2;
static constexpr int S    = 2048;
static constexpr int BS   = Bz * S;      // 4096 tokens
static constexpr int D    = 1024;        // d_model
static constexpr int NH   = 16;
static constexpr int DH   = 64;
static constexpr int DM   = 4096;        // d_mlp

// GEMM tiling: 256 threads = 8 wave32 waves; block tile 128x64, K-step 32.
static constexpr int TBM = 128;
static constexpr int TBN = 64;
static constexpr int TBK = 32;
static constexpr int BKP = 40;           // padded LDS stride (80B rows, 16B-aligned)

#define GF_GELU   1
#define GF_STORET 2

// ---------------------------------------------------------------------------
// Async global->LDS copy (CDNA5 GLOBAL_LOAD_ASYNC_TO_LDS_B128, ASYNCcnt),
// with a compile-safe synchronous fallback if the builtins are absent.
// ---------------------------------------------------------------------------
#if defined(__has_builtin)
#if __has_builtin(__builtin_amdgcn_global_load_async_to_lds_b128) && \
    __has_builtin(__builtin_amdgcn_s_wait_asynccnt)
#define USE_ASYNC_LDS 1
#endif
#endif
#ifndef USE_ASYNC_LDS
#define USE_ASYNC_LDS 0
#endif

__device__ __forceinline__ void copy16_g2l(const _Float16* g, _Float16* l)
{
#if USE_ASYNC_LDS
    // generic LDS pointer: low 32 bits are the LDS byte offset (ISA flat rules);
    // AS(3) pointers are 32-bit on AMDGPU.
    __builtin_amdgcn_global_load_async_to_lds_b128(
        (gptr_v4i)(uintptr_t)g,
        (lptr_v4i)(uint32_t)(uintptr_t)l,
        0, 0);
#else
    *(v8h*)l = *(const v8h*)g;
#endif
}

__device__ __forceinline__ void wait_async_copies()
{
#if USE_ASYNC_LDS
    __builtin_amdgcn_s_wait_asynccnt(0);
#endif
}

// ---------------------------------------------------------------------------
// f16 WMMA GEMM: C(MxN) = alpha * A(MxK) x B(KxN) [+bias][gelu][+addend]
// A row-major f16 (lda).  B ALWAYS stored N-major: Bt is N x K row-major (ldb),
// so every global->LDS transfer is a K-contiguous 16-byte async copy.
// Output f32 (outF) or f16 (outH); GF_STORET stores f16 transposed (N-major,
// leading dim ldc).  Requires M%128 == N%64 == K%32 == 0.
// ---------------------------------------------------------------------------
__global__ __launch_bounds__(256)
void gemm_wmma(const _Float16* __restrict__ A, const _Float16* __restrict__ Bt,
               const float* __restrict__ bias, const float* __restrict__ addend,
               float* __restrict__ outF, _Float16* __restrict__ outH,
               int M, int N, int K, int lda, int ldb, int ldc,
               float alpha, int flags)
{
    __shared__ __attribute__((aligned(16))) _Float16 As[2][TBM][BKP];
    __shared__ __attribute__((aligned(16))) _Float16 Bs[2][TBN][BKP];

    const int tid    = threadIdx.x;
    const int wave   = tid >> 5;
    const int lane   = tid & 31;
    const int wm     = wave & 3;        // wave M offset = wm*32
    const int wn     = wave >> 2;       // wave N offset = wn*32
    const int laneHi = lane >> 4;       // 0/1
    const int laneLo = lane & 15;
    const int blockN = blockIdx.x * TBN;
    const int blockM = blockIdx.y * TBM;

    // staging assignments (fully unrolled, no divergence)
    const int arow = tid >> 2;          // 0..63 (and +64 for second copy)
    const int acol = (tid & 3) << 3;    // 0/8/16/24
    const int brow = tid >> 2;          // 0..63
    const int bcol = (tid & 3) << 3;

    auto stage = [&](int buf, int k0) {
        copy16_g2l(A  + (size_t)(blockM + arow)      * lda + k0 + acol, &As[buf][arow][acol]);
        copy16_g2l(A  + (size_t)(blockM + arow + 64) * lda + k0 + acol, &As[buf][arow + 64][acol]);
        copy16_g2l(Bt + (size_t)(blockN + brow)      * ldb + k0 + bcol, &Bs[buf][brow][bcol]);
    };

    v8f acc[2][2] = {};

    const int nk = K / TBK;
    stage(0, 0);
    for (int kt = 0; kt < nk; ++kt) {
        wait_async_copies();            // this wave's fills of buf (kt&1) done
        __syncthreads();                // everyone's fills visible; prior reads retired
        if (kt + 1 < nk) stage((kt + 1) & 1, (kt + 1) * TBK);
        const int buf = kt & 1;

        // ---- fragments (ISA wave32 layouts) + multiply ----
        v16h afr[2], bfr[2];
#pragma unroll
        for (int im = 0; im < 2; ++im) {
            const int m = wm * 32 + im * 16 + laneLo;
            v8h lo = *(const v8h*)(&As[buf][m][laneHi * 8]);       // K = hi*8 + 0..7
            v8h hi = *(const v8h*)(&As[buf][m][16 + laneHi * 8]);  // K = 16 + hi*8 + 0..7
            afr[im] = __builtin_shufflevector(lo, hi, 0,1,2,3,4,5,6,7,8,9,10,11,12,13,14,15);
        }
#pragma unroll
        for (int in = 0; in < 2; ++in) {
            const int n = wn * 32 + in * 16 + laneLo;
            v8h lo = *(const v8h*)(&Bs[buf][n][laneHi * 8]);       // K = hi*8 + 0..7
            v8h hi = *(const v8h*)(&Bs[buf][n][16 + laneHi * 8]);  // K = 16 + hi*8 + 0..7
            bfr[in] = __builtin_shufflevector(lo, hi, 0,1,2,3,4,5,6,7,8,9,10,11,12,13,14,15);
        }
#pragma unroll
        for (int im = 0; im < 2; ++im)
#pragma unroll
            for (int in = 0; in < 2; ++in)
                acc[im][in] = __builtin_amdgcn_wmma_f32_16x16x32_f16(
                    false, afr[im], false, bfr[in], (short)0, acc[im][in], false, false);
    }

    // ---- epilogue: C VGPR r -> M = r + 8*(lane>>4), N = lane&15 ----
    const bool gelu   = (flags & GF_GELU)   != 0;
    const bool storeT = (flags & GF_STORET) != 0;
#pragma unroll
    for (int im = 0; im < 2; ++im) {
#pragma unroll
        for (int in = 0; in < 2; ++in) {
            const int gN = blockN + wn * 32 + in * 16 + laneLo;
            const float bv = bias ? bias[gN] : 0.0f;
#pragma unroll
            for (int r = 0; r < 8; ++r) {
                const int gM = blockM + wm * 32 + im * 16 + laneHi * 8 + r;
                float v = acc[im][in][r] * alpha + bv;
                if (gelu) v = 0.5f * v * (1.0f + erff(v * 0.70710678118654752f));
                if (addend) v += addend[(size_t)gM * ldc + gN];
                if (outF)        outF[(size_t)gM * ldc + gN] = v;
                else if (storeT) outH[(size_t)gN * ldc + gM] = (_Float16)v;
                else             outH[(size_t)gM * ldc + gN] = (_Float16)v;
            }
        }
    }
}

// ---------------------------------------------------------------------------
// LayerNorm (biased variance, eps=1e-5) over rows of 1024, f32 in -> f16 out
// ---------------------------------------------------------------------------
__global__ __launch_bounds__(256)
void layernorm_f16(const float* __restrict__ x, const float* __restrict__ w,
                   const float* __restrict__ b, _Float16* __restrict__ out)
{
    __shared__ float redS[256];
    __shared__ float redQ[256];
    const int row = blockIdx.x;
    const int tid = threadIdx.x;
    const float* xr = x + (size_t)row * D;
    float vals[4], s = 0.f, sq = 0.f;
#pragma unroll
    for (int i = 0; i < 4; ++i) {
        const float v = xr[tid + (i << 8)];
        vals[i] = v; s += v; sq += v * v;
    }
    redS[tid] = s; redQ[tid] = sq;
    __syncthreads();
    for (int off = 128; off > 0; off >>= 1) {
        if (tid < off) { redS[tid] += redS[tid + off]; redQ[tid] += redQ[tid + off]; }
        __syncthreads();
    }
    const float mean = redS[0] * (1.0f / D);
    const float var  = redQ[0] * (1.0f / D) - mean * mean;
    const float rstd = rsqrtf(var + 1e-5f);
    _Float16* orow = out + (size_t)row * D;
#pragma unroll
    for (int i = 0; i < 4; ++i) {
        const int c = tid + (i << 8);
        orow[c] = (_Float16)((vals[i] - mean) * rstd * w[c] + b[c]);
    }
}

// ---------------------------------------------------------------------------
// Causal softmax, faithful quirk: masked (k > q) scores become 1e-10 (not -inf)
// and still participate in max & sum.  f32 scores in -> f16 probs out.
// ---------------------------------------------------------------------------
__global__ __launch_bounds__(256)
void softmax_causal(const float* __restrict__ s, _Float16* __restrict__ p)
{
    __shared__ float red[256];
    const int q   = blockIdx.x;
    const int tid = threadIdx.x;
    const float* sr = s + (size_t)q * S;
    float vals[8], mx = -3.402823466e38f;
#pragma unroll
    for (int i = 0; i < 8; ++i) {
        const int k = tid + (i << 8);
        const float v = (k > q) ? 1e-10f : sr[k];
        vals[i] = v; mx = fmaxf(mx, v);
    }
    red[tid] = mx;
    __syncthreads();
    for (int off = 128; off > 0; off >>= 1) {
        if (tid < off) red[tid] = fmaxf(red[tid], red[tid + off]);
        __syncthreads();
    }
    const float rowmax = red[0];
    __syncthreads();
    float sum = 0.f;
#pragma unroll
    for (int i = 0; i < 8; ++i) { vals[i] = expf(vals[i] - rowmax); sum += vals[i]; }
    red[tid] = sum;
    __syncthreads();
    for (int off = 128; off > 0; off >>= 1) {
        if (tid < off) red[tid] += red[tid + off];
        __syncthreads();
    }
    const float inv = 1.0f / red[0];
    _Float16* pr = p + (size_t)q * S;
#pragma unroll
    for (int i = 0; i < 8; ++i) pr[tid + (i << 8)] = (_Float16)(vals[i] * inv);
}

// ---------------------------------------------------------------------------
// Weight repack / precision conversion (all outputs N-major = N x K row-major)
// ---------------------------------------------------------------------------
// in: rows x cols f32 row-major; out: cols x rows f16 (transposed)
__global__ __launch_bounds__(256)
void f32_to_f16_t(const float* __restrict__ in, _Float16* __restrict__ out,
                  int rows, int cols)
{
    const int i = blockIdx.x * 256 + threadIdx.x;
    const int r = i / cols, c = i % cols;
    out[(size_t)c * rows + r] = (_Float16)in[i];
}

// (NH, D, DH) -> (NH*DH, D) f16 (N-major) so QKV are plain GEMMs with NxK B
__global__ __launch_bounds__(256)
void repack_head_w_t(const float* __restrict__ in, _Float16* __restrict__ out)
{
    const int i = blockIdx.x * 256 + threadIdx.x;   // over NH*D*DH = 1048576
    const int n = i >> 16;
    const int e = (i >> 6) & 1023;
    const int h = i & 63;
    out[(size_t)(n * DH + h) * D + e] = (_Float16)in[i];
}

// ---------------------------------------------------------------------------
extern "C" void kernel_launch(void* const* d_in, const int* in_sizes, int n_in,
                              void* d_out, int out_size, void* d_ws, size_t ws_size,
                              hipStream_t stream)
{
    const float* residual   = (const float*)d_in[0];
    const float* W_key      = (const float*)d_in[1];
    const float* W_query    = (const float*)d_in[2];
    const float* W_values   = (const float*)d_in[3];
    const float* W_attn_out = (const float*)d_in[4];
    const float* B_key      = (const float*)d_in[5];
    const float* B_query    = (const float*)d_in[6];
    const float* B_values   = (const float*)d_in[7];
    const float* B_attn_out = (const float*)d_in[8];
    const float* ln1_w      = (const float*)d_in[9];
    const float* ln1_b      = (const float*)d_in[10];
    const float* ln2_w      = (const float*)d_in[11];
    const float* ln2_b      = (const float*)d_in[12];
    const float* W_mlp_in   = (const float*)d_in[13];
    const float* W_mlp_out  = (const float*)d_in[14];
    const float* B_mlp_in   = (const float*)d_in[15];
    const float* B_mlp_out  = (const float*)d_in[16];
    float* out = (float*)d_out;

    // workspace carve-out (~150 MB total)
    char* wsp = (char*)d_ws;
    auto carve = [&](size_t bytes) -> char* {
        char* p = wsp; wsp += (bytes + 255) & ~(size_t)255; return p;
    };
    _Float16* xln    = (_Float16*)carve((size_t)BS * D * 2);
    _Float16* xln2   = (_Float16*)carve((size_t)BS * D * 2);
    _Float16* wq_t   = (_Float16*)carve((size_t)D * D * 2);    // (NH*DH) x D
    _Float16* wk_t   = (_Float16*)carve((size_t)D * D * 2);
    _Float16* wv_t   = (_Float16*)carve((size_t)D * D * 2);
    _Float16* wo_t   = (_Float16*)carve((size_t)D * D * 2);    // D x (NH*DH)
    _Float16* wmi_t  = (_Float16*)carve((size_t)D * DM * 2);   // DM x D
    _Float16* wmo_t  = (_Float16*)carve((size_t)DM * D * 2);   // D x DM
    _Float16* Qh     = (_Float16*)carve((size_t)BS * D * 2);   // token-major
    _Float16* Kh     = (_Float16*)carve((size_t)BS * D * 2);   // token-major
    _Float16* Vt     = (_Float16*)carve((size_t)BS * D * 2);   // channel-major (transposed)
    _Float16* Wt     = (_Float16*)carve((size_t)BS * D * 2);   // attn-weighted values
    float*    scores = (float*)   carve((size_t)S * S * 4);    // reused per (b,head)
    _Float16* probs  = (_Float16*)carve((size_t)S * S * 2);    // reused per (b,head)
    float*    mid    = (float*)   carve((size_t)BS * D * 4);
    _Float16* hbuf   = (_Float16*)carve((size_t)BS * DM * 2);

    auto gemm = [&](const _Float16* A, const _Float16* Bt, const float* bias,
                    const float* addend, float* oF, _Float16* oH,
                    int M, int N, int K, int lda, int ldb, int ldc,
                    float alpha, int flags) {
        dim3 grid(N / TBN, M / TBM);
        gemm_wmma<<<grid, 256, 0, stream>>>(A, Bt, bias, addend, oF, oH,
                                            M, N, K, lda, ldb, ldc, alpha, flags);
    };

    // 1) weight repack / conversion (everything to N-major f16)
    repack_head_w_t<<<(NH * D * DH) / 256, 256, 0, stream>>>(W_query,  wq_t);
    repack_head_w_t<<<(NH * D * DH) / 256, 256, 0, stream>>>(W_key,    wk_t);
    repack_head_w_t<<<(NH * D * DH) / 256, 256, 0, stream>>>(W_values, wv_t);
    f32_to_f16_t<<<(D * D)  / 256, 256, 0, stream>>>(W_attn_out, wo_t,  D,  D);
    f32_to_f16_t<<<(D * DM) / 256, 256, 0, stream>>>(W_mlp_in,   wmi_t, D,  DM);
    f32_to_f16_t<<<(DM * D) / 256, 256, 0, stream>>>(W_mlp_out,  wmo_t, DM, D);

    // 2) ln1 -> f16
    layernorm_f16<<<BS, 256, 0, stream>>>(residual, ln1_w, ln1_b, xln);

    // 3) Q, K token-major; V stored channel-major (GF_STORET) for probs x V
    gemm(xln, wq_t, B_query,  nullptr, nullptr, Qh, BS, D, D, D, D, D,  1.0f, 0);
    gemm(xln, wk_t, B_key,    nullptr, nullptr, Kh, BS, D, D, D, D, D,  1.0f, 0);
    gemm(xln, wv_t, B_values, nullptr, nullptr, Vt, BS, D, D, D, D, BS, 1.0f, GF_STORET);

    // 4) attention per (batch, head): scores -> masked softmax -> probs x V
    for (int b = 0; b < Bz; ++b) {
        for (int h = 0; h < NH; ++h) {
            const size_t qkoff = (size_t)b * S * D + (size_t)h * DH;
            const size_t voff  = (size_t)h * DH * BS + (size_t)b * S;
            // scores = (Q x K^T)/8 ; K token-major rows are exactly N-major B
            gemm(Qh + qkoff, Kh + qkoff, nullptr, nullptr, scores, nullptr,
                 S, S, DH, D, D, S, 0.125f, 0);
            softmax_causal<<<S, 256, 0, stream>>>(scores, probs);
            // weighted = probs x V ; V channel-major rows are N-major B
            gemm(probs, Vt + voff, nullptr, nullptr, nullptr, Wt + qkoff,
                 S, DH, S, S, BS, D, 1.0f, 0);
        }
    }

    // 5) attention output projection + residual -> mid
    gemm(Wt, wo_t, B_attn_out, residual, mid, nullptr, BS, D, D, D, D, D, 1.0f, 0);

    // 6) ln2 -> f16
    layernorm_f16<<<BS, 256, 0, stream>>>(mid, ln2_w, ln2_b, xln2);

    // 7) MLP: gelu(x W_in + b) W_out + b + mid
    gemm(xln2, wmi_t, B_mlp_in, nullptr, nullptr, hbuf,
         BS, DM, D, D, D, DM, 1.0f, GF_GELU);
    gemm(hbuf, wmo_t, B_mlp_out, mid, out, nullptr,
         BS, D, DM, DM, DM, D, 1.0f, 0);
}
